// NeuroCore_81801947119870
// MI455X (gfx1250) — compile-verified, matrix operand order checked by reading, not run on hardware
//
#include <hip/hip_runtime.h>
#include <math.h>

#define FM      128
#define NVARS   5120
#define NLITS   10240
#define NCLS    20480
#define NNZT    (NCLS * 3)
#define NG      32
#define NROUNDS 32

typedef __attribute__((ext_vector_type(16))) _Float16 v16h;
typedef __attribute__((ext_vector_type(8)))  float    v8f;
typedef __attribute__((ext_vector_type(4)))  unsigned int u32x4;
typedef __attribute__((ext_vector_type(8)))  unsigned int u32x8;

// ---------------- virtual concat A-operand addressing ----------------
// MODE 0: plain [M x 128]            (A0)
// MODE 1: [C | LC]       K=256       (A0 | A1)
// MODE 2: [L | CL | flip] K=384      (A0 | A1 | A0 flipped)
// MODE 3: [v | vn]       K=256       (A0 rows r and r+NVARS)
template<int MODE>
__device__ __forceinline__ const float* aelem(const float* A0, const float* A1,
                                              int row, int k) {
  if constexpr (MODE == 0) { (void)A1; return A0 + row * FM + k; }
  else if constexpr (MODE == 1) {
    return (k < FM) ? (A0 + row * FM + k) : (A1 + row * FM + (k - FM));
  } else if constexpr (MODE == 2) {
    if (k < FM)     return A0 + row * FM + k;
    if (k < 2 * FM) return A1 + row * FM + (k - FM);
    int fr = (row < NVARS) ? (row + NVARS) : (row - NVARS);
    return A0 + fr * FM + (k - 2 * FM);
  } else {
    (void)A1;
    return (k < FM) ? (A0 + row * FM + k)
                    : (A0 + (row + NVARS) * FM + (k - FM));
  }
}

// ---------------- WMMA GEMM: Out[M x 128] = act(A @ W + b) ----------------
// Block: 256 threads = 8 waves; tile 128 rows x 128 cols.
// Wave w covers rows [16w,16w+16); 8 accumulators of 16x16, K stepped by 32.
// Bsw is pre-swizzled f16 weights in fragment layout [K/32][8][32][16].
// B staging into LDS is done by the Tensor Data Mover (one 1-D descriptor).
template<int KDIM, int MODE, bool RELU>
__global__ __launch_bounds__(256) void gemm_wmma(
    const float* __restrict__ A0, const float* __restrict__ A1,
    const _Float16* __restrict__ Bsw, const float* __restrict__ bias,
    float* __restrict__ Out)
{
  constexpr int NSTEP = KDIM / 32;
  constexpr int SLDS  = (NSTEP > 8) ? 8 : NSTEP;   // stage <=64KB of B in LDS
  extern __shared__ _Float16 Blds[];

  { // TDM: stage SLDS*8192 bytes of swizzled B from global into LDS.
    constexpr unsigned NE8 = (unsigned)SLDS * 1024u;  // 8-byte elements
    unsigned lds_off = (unsigned)(size_t)(&Blds[0]);
    unsigned long long ga = (unsigned long long)(size_t)Bsw;
    u32x4 g0;
    g0[0] = 1u;                                  // count=1, user descriptor
    g0[1] = lds_off;                             // lds_addr
    g0[2] = (unsigned)ga;                        // global_addr[31:0]
    g0[3] = (unsigned)(ga >> 32) | (2u << 30);   // global_addr[56:32] | type=2
    u32x8 g1;
    g1[0] = 3u << 16;                            // workgroup_mask=0, data_size=8B
    g1[1] = (NE8 & 0xffffu) << 16;               // tensor_dim0[15:0]
    g1[2] = (NE8 >> 16) | (1u << 16);            // tensor_dim0[31:16] | tensor_dim1=1
    g1[3] = (NE8 & 0xffffu) << 16;               // tile_dim0 = NE8
    g1[4] = 1u;                                  // tile_dim1=1, tile_dim2=0
    g1[5] = NE8;                                 // tensor_dim0_stride[31:0]
    g1[6] = 0u;                                  // stride0 hi | stride1 lo
    g1[7] = 0u;                                  // stride1 hi
    if (threadIdx.x < 32) {                      // wave 0 issues the DMA
      asm volatile("tensor_load_to_lds %0, %1" :: "s"(g0), "s"(g1) : "memory");
    }
    __builtin_amdgcn_s_wait_tensorcnt(0);
  }
  __syncthreads();

  const int w     = threadIdx.x >> 5;
  const int l     = threadIdx.x & 31;
  const int row0  = blockIdx.x * 128;
  const int arow  = row0 + w * 16 + (l & 15);
  const int kbase = (l >= 16) ? 8 : 0;

  __builtin_prefetch(aelem<MODE>(A0, A1, arow, kbase), 0, 1);

  v8f acc[8] = {};

#pragma unroll
  for (int s = 0; s < NSTEP; ++s) {
    const int k0 = s * 32;
    // A fragment: lane holds 16 f16; two contiguous 8-float runs
    float fa[16];
    {
      const float4* p = (const float4*)aelem<MODE>(A0, A1, arow, k0 + kbase);
      float4 x0 = p[0], x1 = p[1];
      const float4* q = (const float4*)aelem<MODE>(A0, A1, arow, k0 + kbase + 16);
      float4 y0 = q[0], y1 = q[1];
      fa[0]=x0.x; fa[1]=x0.y; fa[2]=x0.z; fa[3]=x0.w;
      fa[4]=x1.x; fa[5]=x1.y; fa[6]=x1.z; fa[7]=x1.w;
      fa[8]=y0.x; fa[9]=y0.y; fa[10]=y0.z; fa[11]=y0.w;
      fa[12]=y1.x; fa[13]=y1.y; fa[14]=y1.z; fa[15]=y1.w;
    }
    v16h a;
#pragma unroll
    for (int i = 0; i < 16; ++i) a[i] = (_Float16)fa[i];

    if (s < SLDS) {
#pragma unroll
      for (int j = 0; j < 8; ++j) {
        v16h b = *(const v16h*)(Blds + (((s * 8 + j) * 32 + l) * 16));
        acc[j] = __builtin_amdgcn_wmma_f32_16x16x32_f16(
            false, a, false, b, (short)0, acc[j], false, false);
      }
    } else {
#pragma unroll
      for (int j = 0; j < 8; ++j) {
        v16h b = *(const v16h*)(Bsw + (((size_t)(s * 8 + j) * 32 + l) * 16));
        acc[j] = __builtin_amdgcn_wmma_f32_16x16x32_f16(
            false, a, false, b, (short)0, acc[j], false, false);
      }
    }
  }

  // C/D layout: lanes 0-15 -> N=lane, VGPR v -> M=v; lanes 16-31 -> M=v+8
  const int orow = row0 + w * 16 + ((l >= 16) ? 8 : 0);
  const int c0   = l & 15;
#pragma unroll
  for (int j = 0; j < 8; ++j) {
    const int col = j * 16 + c0;
    const float bv = bias[col];
#pragma unroll
    for (int v = 0; v < 8; ++v) {
      float x = acc[j][v] + bv;
      if (RELU) x = fminf(fmaxf(x, 0.f), 6.f);
      Out[(size_t)(orow + v) * FM + col] = x;
    }
  }
}

// -------- weight convert + swizzle into WMMA B-fragment layout --------
__global__ void k_cvtw(const float* __restrict__ W, _Float16* __restrict__ out,
                       int kdim) {
  int total = kdim * FM;
  for (int g = blockIdx.x * blockDim.x + threadIdx.x; g < total;
       g += gridDim.x * blockDim.x) {
    int i = g & 15, l = (g >> 4) & 31, j = (g >> 9) & 7, s = g >> 12;
    int kk = ((i < 8) ? i : i + 8) + ((l >= 16) ? 8 : 0);
    int k = s * 32 + kk;
    int n = j * 16 + (l & 15);
    out[g] = (_Float16)W[k * FM + n];
  }
}

__global__ void k_fill(float* __restrict__ p, int n, const float* __restrict__ v) {
  int t = blockIdx.x * blockDim.x + threadIdx.x;
  if (t < n) p[t] = v[0];
}

// LC[c,f] = LC_scale * sum_k L[lit[3c+k], f] * vals[3c+k]   (structured gather)
__global__ void k_lc(const float* __restrict__ L, const int* __restrict__ lit,
                     const float* __restrict__ vals, const float* __restrict__ sc,
                     float* __restrict__ LC) {
  int t = blockIdx.x * 256 + threadIdx.x;
  if (t >= NCLS * FM) return;
  int c = t >> 7, f = t & 127, e = 3 * c;
  float s = L[(size_t)lit[e] * FM + f] * vals[e]
          + L[(size_t)lit[e + 1] * FM + f] * vals[e + 1]
          + L[(size_t)lit[e + 2] * FM + f] * vals[e + 2];
  LC[t] = s * sc[0];
}

// CL scatter-add (global_atomic_add_f32), CL pre-zeroed
__global__ void k_cl(const float* __restrict__ C, const int* __restrict__ lit,
                     const int* __restrict__ cls, const float* __restrict__ vals,
                     const float* __restrict__ sc, float* __restrict__ CL) {
  int t = blockIdx.x * 256 + threadIdx.x;
  if (t >= NNZT * FM) return;
  int e = t >> 7, f = t & 127;
  atomicAdd(&CL[(size_t)lit[e] * FM + f],
            C[(size_t)cls[e] * FM + f] * vals[e] * sc[0]);
}

// per-column (axis 0) sum / sumsq into stats[0..127] / stats[128..255]
__global__ void k_stats(const float* __restrict__ X, int M, float* __restrict__ st) {
  int f = threadIdx.x & 127;
  int part = blockIdx.x * 2 + (threadIdx.x >> 7);
  int stride = gridDim.x * 2;
  float s = 0.f, s2 = 0.f;
  for (int r = part; r < M; r += stride) {
    float x = X[(size_t)r * FM + f];
    s += x; s2 += x * x;
  }
  atomicAdd(&st[f], s);
  atomicAdd(&st[128 + f], s2);
}

__global__ void k_norm(const float* __restrict__ X, const float* __restrict__ st,
                       float invM, float* __restrict__ Y, int total) {
  int t = blockIdx.x * 256 + threadIdx.x;
  if (t >= total) return;
  int f = t & 127;
  float m = st[f] * invM;
  float v = st[128 + f] * invM - m * m;
  Y[t] = (X[t] - m) * rsqrtf(v + 1e-3f);
}

// logits[r] = dot(H[r], W4) + b4   (one wave per row)
__global__ void k_logits(const float* __restrict__ H, const float* __restrict__ W4,
                         const float* __restrict__ b4, float* __restrict__ lg) {
  int r = blockIdx.x, l = threadIdx.x;
  float s = 0.f;
  for (int k = l; k < FM; k += 32) s += H[(size_t)r * FM + k] * W4[k];
  for (int off = 16; off; off >>= 1) s += __shfl_xor(s, off, 32);
  if (l == 0) lg[r] = s + b4[0];
}

// per-clause penalty -> per-graph sums (pg pre-zeroed)
__global__ void k_cv(const float* __restrict__ lg, const int* __restrict__ lit,
                     const int* __restrict__ cgid, const float* __restrict__ vals,
                     float* __restrict__ pg) {
  int c = blockIdx.x * 256 + threadIdx.x;
  if (c >= NCLS) return;
  float cv = 0.f;
#pragma unroll
  for (int k = 0; k < 3; ++k) {
    int e = 3 * c + k;
    int li = lit[e];
    float x = (li < NVARS) ? lg[li] : -lg[li - NVARS];
    float sp = fmaxf(x, 0.f) + log1pf(expf(-fabsf(x)));  // softplus
    cv += sp * vals[e];
  }
  float ce = expf(-cv);
  float pc = ce * (-logf(1.0f - ce + 1e-8f));
  atomicAdd(&pg[cgid[c]], pc);
}

// loss += sum_g sqrt(pg+1e-6)-sqrt(1e-6)   (cost-weighted sort collapses to sum)
__global__ void k_loss(const float* __restrict__ pg, float* __restrict__ acc) {
  int g = threadIdx.x;
  float v = sqrtf(pg[g] + 1e-6f) - sqrtf(1e-6f);
  for (int off = 16; off; off >>= 1) v += __shfl_xor(v, off, 32);
  if (g == 0) acc[0] += v;
}

// out = [final logits (best_var == 0), loss/ROUNDS]
__global__ void k_out(const float* __restrict__ lg, const float* __restrict__ acc,
                      float* __restrict__ out) {
  int t = blockIdx.x * blockDim.x + threadIdx.x;
  if (t < NVARS) out[t] = lg[t];
  if (t == 0) out[NVARS] = acc[0] / (float)NROUNDS;
}

extern "C" void kernel_launch(void* const* d_in, const int* in_sizes, int n_in,
                              void* d_out, int out_size, void* d_ws, size_t ws_size,
                              hipStream_t stream) {
  (void)in_sizes; (void)n_in; (void)out_size; (void)ws_size;

  const int*   lit  = (const int*)d_in[0];
  const int*   cls  = (const int*)d_in[1];
  const int*   cgid = (const int*)d_in[2];
  const float* vals = (const float*)d_in[4];
  const float* Lw1 = (const float*)d_in[5];  const float* Lb1 = (const float*)d_in[6];
  const float* Lw2 = (const float*)d_in[7];  const float* Lb2 = (const float*)d_in[8];
  const float* Cw1 = (const float*)d_in[9];  const float* Cb1 = (const float*)d_in[10];
  const float* Cw2 = (const float*)d_in[11]; const float* Cb2 = (const float*)d_in[12];
  const float* Vw1 = (const float*)d_in[13]; const float* Vb1 = (const float*)d_in[14];
  const float* Vw2 = (const float*)d_in[15]; const float* Vb2 = (const float*)d_in[16];
  const float* Vw3 = (const float*)d_in[17]; const float* Vb3 = (const float*)d_in[18];
  const float* Vw4 = (const float*)d_in[19]; const float* Vb4 = (const float*)d_in[20];
  const float* Lis = (const float*)d_in[21]; const float* Cis = (const float*)d_in[22];
  const float* LCs = (const float*)d_in[23]; const float* CLs = (const float*)d_in[24];

  float* F = (float*)d_ws;
  size_t o = 0;
  auto take = [&](size_t n) { size_t r = o; o += (n + 63) & ~(size_t)63; return r; };
  float* L    = F + take((size_t)NLITS * FM);
  float* C    = F + take((size_t)NCLS * FM);
  float* LC   = F + take((size_t)NCLS * FM);
  float* CL   = F + take((size_t)NLITS * FM);
  float* T1   = F + take((size_t)NCLS * FM);
  float* T2   = F + take((size_t)NCLS * FM);
  float* lgts = F + take(NVARS);
  float* pg   = F + take(64);
  float* lacc = F + take(64);
  float* stat = F + take(256);

  _Float16* H = (_Float16*)(F + o);
  size_t ho = 0;
  auto takeh = [&](size_t n) { size_t r = ho; ho += (n + 63) & ~(size_t)63; return r; };
  _Float16* Cw1h = H + takeh(256 * FM);
  _Float16* Cw2h = H + takeh(128 * FM);
  _Float16* Lw1h = H + takeh(384 * FM);
  _Float16* Lw2h = H + takeh(128 * FM);
  _Float16* Vw1h = H + takeh(256 * FM);
  _Float16* Vw2h = H + takeh(128 * FM);
  _Float16* Vw3h = H + takeh(128 * FM);

  // one-time per launch: convert+swizzle weights, init state
  k_cvtw<<<128, 256, 0, stream>>>(Cw1, Cw1h, 256);
  k_cvtw<<<64,  256, 0, stream>>>(Cw2, Cw2h, 128);
  k_cvtw<<<192, 256, 0, stream>>>(Lw1, Lw1h, 384);
  k_cvtw<<<64,  256, 0, stream>>>(Lw2, Lw2h, 128);
  k_cvtw<<<128, 256, 0, stream>>>(Vw1, Vw1h, 256);
  k_cvtw<<<64,  256, 0, stream>>>(Vw2, Vw2h, 128);
  k_cvtw<<<64,  256, 0, stream>>>(Vw3, Vw3h, 128);
  k_fill<<<(NLITS * FM + 255) / 256, 256, 0, stream>>>(L, NLITS * FM, Lis);
  k_fill<<<(NCLS * FM + 255) / 256, 256, 0, stream>>>(C, NCLS * FM, Cis);
  hipMemsetAsync(lacc, 0, sizeof(float), stream);

  for (int r = 0; r < NROUNDS; ++r) {
    // clause update
    k_lc<<<NCLS * FM / 256, 256, 0, stream>>>(L, lit, vals, LCs, LC);
    gemm_wmma<256, 1, true ><<<NCLS / 128, 256, 65536, stream>>>(C, LC, Cw1h, Cb1, T1);
    gemm_wmma<128, 0, false><<<NCLS / 128, 256, 32768, stream>>>(T1, nullptr, Cw2h, Cb2, T2);
    hipMemsetAsync(stat, 0, 256 * sizeof(float), stream);
    k_stats<<<64, 256, 0, stream>>>(T2, NCLS, stat);
    k_norm<<<NCLS * FM / 256, 256, 0, stream>>>(T2, stat, 1.0f / NCLS, C, NCLS * FM);
    // literal update
    hipMemsetAsync(CL, 0, (size_t)NLITS * FM * sizeof(float), stream);
    k_cl<<<NNZT * FM / 256, 256, 0, stream>>>(C, lit, cls, vals, CLs, CL);
    gemm_wmma<384, 2, true ><<<NLITS / 128, 256, 65536, stream>>>(L, CL, Lw1h, Lb1, T1);
    gemm_wmma<128, 0, false><<<NLITS / 128, 256, 32768, stream>>>(T1, nullptr, Lw2h, Lb2, T2);
    hipMemsetAsync(stat, 0, 256 * sizeof(float), stream);
    k_stats<<<64, 256, 0, stream>>>(T2, NLITS, stat);
    k_norm<<<NLITS * FM / 256, 256, 0, stream>>>(T2, stat, 1.0f / NLITS, L, NLITS * FM);
    // voting head
    gemm_wmma<256, 3, true><<<NVARS / 128, 256, 65536, stream>>>(L, nullptr, Vw1h, Vb1, T1);
    gemm_wmma<128, 0, true><<<NVARS / 128, 256, 32768, stream>>>(T1, nullptr, Vw2h, Vb2, T2);
    gemm_wmma<128, 0, true><<<NVARS / 128, 256, 32768, stream>>>(T2, nullptr, Vw3h, Vb3, T1);
    k_logits<<<NVARS, 32, 0, stream>>>(T1, Vw4, Vb4, lgts);
    // loss
    hipMemsetAsync(pg, 0, NG * sizeof(float), stream);
    k_cv<<<NCLS / 256, 256, 0, stream>>>(lgts, lit, cgid, vals, pg);
    k_loss<<<1, 32, 0, stream>>>(pg, lacc);
  }
  k_out<<<(NVARS + 255) / 256, 256, 0, stream>>>(lgts, lacc, (float*)d_out);
}
